// SA_22505628631351
// MI455X (gfx1250) — matrix operand order repaired and checked by reading, not compiled, not run on hardware
//
#include <hip/hip_runtime.h>
#include <hip/hip_bf16.h>
#include <math.h>

// ---------------------------------------------------------------------------
// Types for CDNA5 WMMA (wave32)
// ---------------------------------------------------------------------------
typedef __bf16 bf16_t;
typedef __attribute__((ext_vector_type(16))) __bf16 v16bf;
typedef __attribute__((ext_vector_type(8)))  __bf16 v8bf;
typedef __attribute__((ext_vector_type(8)))  float  v8f;

__device__ __forceinline__ v8f wmma_bf16(v16bf a, v16bf b, v8f c) {
    // D = A(16x32 bf16) * B(32x16 bf16) + C(16x16 f32)
    return __builtin_amdgcn_wmma_f32_16x16x32_bf16(
        /*neg_a=*/false, a, /*neg_b=*/false, b,
        /*c_mod=*/(short)0, c, /*reuse_a=*/false, /*reuse_b=*/false);
}

// float -> bf16 round-to-nearest-even via bit manipulation
__device__ __forceinline__ bf16_t f2bf(float f) {
    union { float f; unsigned u; } x; x.f = f;
    unsigned r = x.u + 0x7FFFu + ((x.u >> 16) & 1u);
    unsigned short h = (unsigned short)(r >> 16);
    return __builtin_bit_cast(bf16_t, h);
}

// ---------------------------------------------------------------------------
// gfx1250 async global->LDS copy (ASYNCcnt path), with safe fallback.
// Probe-learned signature: (as1 int4* gsrc, as3 int4* ldst, imm off, imm cpol)
// ---------------------------------------------------------------------------
#if __has_builtin(__builtin_amdgcn_global_load_async_to_lds_b128)
#define HAVE_ASYNC_LDS 1
#else
#define HAVE_ASYNC_LDS 0
#endif

typedef int v4i_ __attribute__((vector_size(16)));
typedef __attribute__((address_space(1))) v4i_* g_v4i_ptr;
typedef __attribute__((address_space(3))) v4i_* l_v4i_ptr;

__device__ __forceinline__ void async_g2l_b128(const bf16_t* g, bf16_t* l) {
#if HAVE_ASYNC_LDS
    __builtin_amdgcn_global_load_async_to_lds_b128(
        (g_v4i_ptr)(g), (l_v4i_ptr)(l), 0, 0);
#else
    *(v8bf*)l = *(const v8bf*)g;          // synchronous 16B fallback
#endif
}

__device__ __forceinline__ void wait_async0() {
#if HAVE_ASYNC_LDS
#if __has_builtin(__builtin_amdgcn_s_wait_asynccnt)
    __builtin_amdgcn_s_wait_asynccnt(0);
#else
    asm volatile("s_wait_asynccnt 0x0" ::: "memory");
#endif
#endif
}

// ---------------------------------------------------------------------------
// Tile loaders
// ---------------------------------------------------------------------------
// A-matrix tile (16x32 bf16), row-major src[row*ld + k].
// Lane L<16: row=L, elems = K{0..7, 16..23}; lane L>=16: row=L-16, K{8..15, 24..31}.
__device__ __forceinline__ v16bf load_A(const bf16_t* base, int ld) {
    const int lane = threadIdx.x & 31;
    const int row  = lane & 15;
    const int koff = (lane >> 4) << 3;           // 0 or 8
    const bf16_t* p = base + row * ld + koff;
    v8bf lo = *(const v8bf*)(p);
    v8bf hi = *(const v8bf*)(p + 16);
    return __builtin_shufflevector(lo, hi, 0,1,2,3,4,5,6,7,8,9,10,11,12,13,14,15);
}

// B-matrix tile (32x16 bf16), "N-major" memory: src[n*ld + k] == B(K=k, N=n).
// Lane L<16: K{0..7, 16..23}; lane L>=16: K{8..15, 24..31} (two 16B loads).
__device__ __forceinline__ v16bf load_B(const bf16_t* base, int ld) {
    const int lane = threadIdx.x & 31;
    const int n    = lane & 15;
    const int koff = (lane >> 4) << 3;           // 0 or 8
    const bf16_t* p = base + n * ld + koff;
    v8bf lo = *(const v8bf*)(p);
    v8bf hi = *(const v8bf*)(p + 16);
    return __builtin_shufflevector(lo, hi, 0,1,2,3,4,5,6,7,8,9,10,11,12,13,14,15);
}

// B-matrix tile, 16B-aligned two-part version (for padded LDS strides).
__device__ __forceinline__ v16bf load_B2(const bf16_t* base, int ld) {
    const int lane = threadIdx.x & 31;
    const int n    = lane & 15;
    const int koff = (lane >> 4) << 3;           // 0 or 8
    const bf16_t* p = base + n * ld + koff;
    v8bf lo = *(const v8bf*)(p);
    v8bf hi = *(const v8bf*)(p + 16);
    return __builtin_shufflevector(lo, hi, 0,1,2,3,4,5,6,7,8,9,10,11,12,13,14,15);
}

// ---------------------------------------------------------------------------
// Problem constants
// ---------------------------------------------------------------------------
#define BB   16
#define NN   2048
#define DE   256
#define DA   64
#define ROWS (BB * NN)        // 32768
#define BN_EPS_F 1e-5f

#define VPAD 40               // LDS row stride for V tile (32 keys + pad)
#define KPAD 72               // LDS row stride for K tile (64 d + pad)
#define WPAD 264              // LDS row stride for Wl panel (256 k + pad)

// ---------------------------------------------------------------------------
// Kernel 0a: x (f32) -> x_bf (bf16)
// ---------------------------------------------------------------------------
__global__ __launch_bounds__(256) void k_cvt_x(const float* __restrict__ x,
                                               bf16_t* __restrict__ xb) {
    int idx = (blockIdx.x * 256 + threadIdx.x) * 4;
    float4 v = *(const float4*)(x + idx);
    xb[idx + 0] = f2bf(v.x);
    xb[idx + 1] = f2bf(v.y);
    xb[idx + 2] = f2bf(v.z);
    xb[idx + 3] = f2bf(v.w);
}

// ---------------------------------------------------------------------------
// Kernel 0b: W [256, dout] f32 -> W^T [dout, 256] bf16
// ---------------------------------------------------------------------------
__global__ __launch_bounds__(256) void k_cvt_wT(const float* __restrict__ w,
                                                bf16_t* __restrict__ wT, int dout) {
    int i = blockIdx.x * 256 + threadIdx.x;
    if (i >= 256 * dout) return;
    int d = i / dout, c = i % dout;
    wT[c * 256 + d] = f2bf(w[i]);
}

// ---------------------------------------------------------------------------
// Kernel 1: fused QKV projection. One wave per 16x16 output tile.
// q,k row-major [row,64] bf16; v stored TRANSPOSED vT[(b*256+ch)*2048 + n].
// ---------------------------------------------------------------------------
__global__ __launch_bounds__(256) void k_qkv(
    const bf16_t* __restrict__ xb,
    const bf16_t* __restrict__ wqT, const bf16_t* __restrict__ wkT,
    const bf16_t* __restrict__ wvT,
    const float* __restrict__ bq, const float* __restrict__ bk,
    const float* __restrict__ bv,
    bf16_t* __restrict__ qb, bf16_t* __restrict__ kb, bf16_t* __restrict__ vT) {
    int wid = blockIdx.x * 8 + (threadIdx.x >> 5);   // 0..49151
    int rt = wid / 24, ct = wid % 24;
    int row0 = rt * 16;
    const int lane = threadIdx.x & 31;

    const bf16_t* wT; const float* bias; int col0;
    if (ct < 4)      { wT = wqT; bias = bq; col0 = ct * 16; }
    else if (ct < 8) { wT = wkT; bias = bk; col0 = (ct - 4) * 16; }
    else             { wT = wvT; bias = bv; col0 = (ct - 8) * 16; }

    v8f acc = {};
    #pragma unroll
    for (int k0 = 0; k0 < DE; k0 += 32) {
        v16bf a = load_A(xb + row0 * DE + k0, DE);
        v16bf b = load_B(wT + col0 * DE + k0, DE);
        acc = wmma_bf16(a, b, acc);
    }
    int c = col0 + (lane & 15);
    float bia = bias[c];
    int mbase = (lane >> 4) * 8;                     // C layout: M = r + mbase
    if (ct < 8) {
        bf16_t* out = (ct < 4) ? qb : kb;
        #pragma unroll
        for (int r = 0; r < 8; ++r)
            out[(row0 + mbase + r) * DA + c] = f2bf(acc[r] + bia);
    } else {
        int bidx = row0 >> 11;
        int n0   = row0 & (NN - 1);
        #pragma unroll
        for (int r = 0; r < 8; ++r)
            vT[(bidx * DE + c) * NN + (n0 + mbase + r)] = f2bf(acc[r] + bia);
    }
}

// ---------------------------------------------------------------------------
// Kernel 2: flash attention. One wave per (batch, 16-query tile); 8 waves of a
// block share one batch and walk keys in lockstep, so K (4KB) and V (16KB) of
// each 32-key step are staged once per block into LDS with double-buffered
// async global->LDS copies (ASYNCcnt), then read back via ds_load.
// S^T = K*Q^T so softmaxed P lands directly in A-matrix layout.
// ---------------------------------------------------------------------------
__global__ __launch_bounds__(256) void k_attn(
    const bf16_t* __restrict__ qb, const bf16_t* __restrict__ kb,
    const bf16_t* __restrict__ vT, bf16_t* __restrict__ attn) {
    __shared__ __align__(16) bf16_t vlds[2][DE * VPAD];   // 2 x 20KB
    __shared__ __align__(16) bf16_t klds[2][32 * KPAD];   // 2 x 4.5KB

    int wid = blockIdx.x * 8 + (threadIdx.x >> 5);   // 0..2047
    int b  = wid >> 7;                               // uniform per block
    int q0 = (wid & 127) * 16;
    const int lane  = threadIdx.x & 31;
    const int mbase = (lane >> 4) * 8;

    // stage one 32-key step (V: 256ch x 32keys, K: 32keys x 64d) into buffer p
    auto stage = [&](int p, int s) {
        const int t = threadIdx.x;                   // 0..255 == V channel
        const bf16_t* vsrc = vT + ((size_t)(b * DE + t) * NN + s);
        #pragma unroll
        for (int part = 0; part < 4; ++part)
            async_g2l_b128(vsrc + part * 8, &vlds[p][t * VPAD + part * 8]);
        int row = t >> 3, part = t & 7;
        async_g2l_b128(kb + ((size_t)(b * NN + s + row) * DA + part * 8),
                       &klds[p][row * KPAD + part * 8]);
    };

    // Q as B-operand (K = d), resident for whole kernel
    const bf16_t* qbase = qb + (b * NN + q0) * DA;
    v16bf qB0 = load_B(qbase + 0,  DA);              // d 0..31
    v16bf qB1 = load_B(qbase + 32, DA);              // d 32..63

    float m = -INFINITY, l = 0.0f;
    v8f o[16];
    #pragma unroll
    for (int i = 0; i < 16; ++i) o[i] = (v8f){};

    const float scl = 1.0f / 64.0f;                  // reference scales by d_a

    stage(0, 0);
    for (int i = 0; i < NN / 32; ++i) {
        int p = i & 1;
        wait_async0();
        __syncthreads();                             // buffer p ready for all
        if (i + 1 < NN / 32) stage(1 - p, (i + 1) * 32);

        const bf16_t* kl = klds[p];
        v16bf ka0 = load_A(kl,                 KPAD);   // keys 0..15,  d 0..31
        v16bf ka1 = load_A(kl + 32,            KPAD);   //              d 32..63
        v16bf kc0 = load_A(kl + 16 * KPAD,      KPAD);  // keys 16..31
        v16bf kc1 = load_A(kl + 16 * KPAD + 32, KPAD);
        v8f z = {};
        v8f stlo = wmma_bf16(ka0, qB0, z); stlo = wmma_bf16(ka1, qB1, stlo);
        v8f sthi = wmma_bf16(kc0, qB0, z); sthi = wmma_bf16(kc1, qB1, sthi);

        // per-lane: 16 score values of one query row (half of the 32 keys)
        float sv[16];
        float mx = -INFINITY;
        #pragma unroll
        for (int r = 0; r < 8; ++r) { sv[r]   = stlo[r] * scl; mx = fmaxf(mx, sv[r]);   }
        #pragma unroll
        for (int r = 0; r < 8; ++r) { sv[8+r] = sthi[r] * scl; mx = fmaxf(mx, sv[8+r]); }
        mx = fmaxf(mx, __shfl_xor(mx, 16, 32));      // combine half-wave partner
        float mnew = fmaxf(m, mx);
        float corr = __expf(m - mnew);               // first iter: exp(-inf)=0

        float ssum = 0.0f;
        v16bf P;
        #pragma unroll
        for (int j = 0; j < 16; ++j) {
            float pv = __expf(sv[j] - mnew);
            ssum += pv;
            P[j] = f2bf(pv);                         // already in A layout
        }
        ssum += __shfl_xor(ssum, 16, 32);
        l = l * corr + ssum;
        m = mnew;

        // broadcast per-query rescale factors into C-tile M indexing
        float cr[8];
        #pragma unroll
        for (int r = 0; r < 8; ++r) cr[r] = __shfl(corr, mbase + r, 32);

        #pragma unroll
        for (int ct = 0; ct < 16; ++ct) {
            v8f oc = o[ct];
            #pragma unroll
            for (int r = 0; r < 8; ++r) oc[r] *= cr[r];
            v16bf vB = load_B2(&vlds[p][(ct * 16) * VPAD], VPAD);
            o[ct] = wmma_bf16(P, vB, oc);
        }
    }

    // epilogue: divide by row sums, store attn row-major bf16
    float rl[8];
    #pragma unroll
    for (int r = 0; r < 8; ++r) {
        float lr = __shfl(l, mbase + r, 32);
        rl[r] = 1.0f / lr;
    }
    bf16_t* obase = attn + (b * NN + q0) * DE;
    int c = lane & 15;
    #pragma unroll
    for (int ct = 0; ct < 16; ++ct)
        #pragma unroll
        for (int r = 0; r < 8; ++r)
            obase[(mbase + r) * DE + ct * 16 + c] = f2bf(o[ct][r] * rl[r]);
}

// ---------------------------------------------------------------------------
// Kernel 3: h = attn @ Wl + bl (WMMA) + per-channel sum/sumsq for BN stats.
// The 8 waves of a block share one 16-column Wl panel: stage it once in LDS
// via async copies; LDS-reduce partial sums before global atomics.
// ---------------------------------------------------------------------------
__global__ __launch_bounds__(256) void k_lin(
    const bf16_t* __restrict__ attn, const bf16_t* __restrict__ wlT,
    const float* __restrict__ bl, float* __restrict__ h,
    float* __restrict__ sums) {
    __shared__ __align__(16) bf16_t wlds[16 * WPAD];  // 8.25KB panel
    __shared__ float r1[8][16];
    __shared__ float r2[8][16];
    int w   = threadIdx.x >> 5;
    int wid = blockIdx.x * 8 + w;                    // 0..32767
    int ct = wid >> 11;                              // uniform per block
    int rt = wid & 2047;
    int row0 = rt * 16, col0 = ct * 16;
    const int lane = threadIdx.x & 31;

    {   // stage Wl columns [col0, col0+16) x 256 once per block
        const int t = threadIdx.x;
        #pragma unroll
        for (int j = 0; j < 2; ++j) {
            int cidx = t * 2 + j;                    // 0..511
            int n = cidx >> 5, part = cidx & 31;
            async_g2l_b128(wlT + (size_t)(col0 + n) * DE + part * 8,
                           &wlds[n * WPAD + part * 8]);
        }
        wait_async0();
        __syncthreads();
    }

    v8f acc = {};
    #pragma unroll
    for (int k0 = 0; k0 < DE; k0 += 32) {
        v16bf a = load_A(attn + row0 * DE + k0, DE);
        v16bf b = load_B2(wlds + k0, WPAD);
        acc = wmma_bf16(a, b, acc);
    }
    int c = col0 + (lane & 15);
    float bia = bl[c];
    int mbase = (lane >> 4) * 8;
    float s1 = 0.0f, s2 = 0.0f;
    #pragma unroll
    for (int r = 0; r < 8; ++r) {
        float v = acc[r] + bia;
        h[(row0 + mbase + r) * DE + c] = v;
        s1 += v; s2 += v * v;
    }
    s1 += __shfl_xor(s1, 16, 32);
    s2 += __shfl_xor(s2, 16, 32);
    if (lane < 16) { r1[w][lane] = s1; r2[w][lane] = s2; }
    __syncthreads();
    if (threadIdx.x < 16) {
        float a1 = 0.0f, a2 = 0.0f;
        #pragma unroll
        for (int ww = 0; ww < 8; ++ww) { a1 += r1[ww][threadIdx.x]; a2 += r2[ww][threadIdx.x]; }
        atomicAdd(&sums[col0 + threadIdx.x],      a1);
        atomicAdd(&sums[DE + col0 + threadIdx.x], a2);
    }
}

// ---------------------------------------------------------------------------
// Kernel 4: finalize BN stats into per-channel scale/shift
// ---------------------------------------------------------------------------
__global__ void k_stats(const float* __restrict__ sums,
                        const float* __restrict__ gamma,
                        const float* __restrict__ beta,
                        float* __restrict__ ss) {
    int c = threadIdx.x;                             // 256 threads
    const float inv = 1.0f / (float)ROWS;
    float mean = sums[c] * inv;
    float var  = sums[DE + c] * inv - mean * mean;
    float sc   = rsqrtf(var + BN_EPS_F) * gamma[c];
    ss[c]      = sc;
    ss[DE + c] = beta[c] - mean * sc;
}

// ---------------------------------------------------------------------------
// Kernel 5: y = relu(h*scale + shift) + x, vectorized float4
// ---------------------------------------------------------------------------
__global__ __launch_bounds__(256) void k_bnres(
    const float* __restrict__ h, const float* __restrict__ x,
    const float* __restrict__ ss, float* __restrict__ y) {
    int idx = (blockIdx.x * 256 + threadIdx.x) * 4;
    float4 hv = *(const float4*)(h + idx);
    float4 xv = *(const float4*)(x + idx);
    int c = idx & (DE - 1);
    float4 r;
    r.x = fmaxf(hv.x * ss[c+0] + ss[DE+c+0], 0.0f) + xv.x;
    r.y = fmaxf(hv.y * ss[c+1] + ss[DE+c+1], 0.0f) + xv.y;
    r.z = fmaxf(hv.z * ss[c+2] + ss[DE+c+2], 0.0f) + xv.z;
    r.w = fmaxf(hv.w * ss[c+3] + ss[DE+c+3], 0.0f) + xv.w;
    *(float4*)(y + idx) = r;
}

// ---------------------------------------------------------------------------
// Host launcher
// ---------------------------------------------------------------------------
extern "C" void kernel_launch(void* const* d_in, const int* in_sizes, int n_in,
                              void* d_out, int out_size, void* d_ws, size_t ws_size,
                              hipStream_t stream) {
    const float* x     = (const float*)d_in[0];
    const float* Wq    = (const float*)d_in[1];
    const float* bq    = (const float*)d_in[2];
    const float* Wk    = (const float*)d_in[3];
    const float* bk    = (const float*)d_in[4];
    const float* Wv    = (const float*)d_in[5];
    const float* bv    = (const float*)d_in[6];
    const float* Wl    = (const float*)d_in[7];
    const float* bl    = (const float*)d_in[8];
    const float* gamma = (const float*)d_in[9];
    const float* beta  = (const float*)d_in[10];
    float* y = (float*)d_out;

    char* ws = (char*)d_ws;
    size_t off = 0;
    auto alloc = [&](size_t bytes) { char* p = ws + off; off = (off + bytes + 255) & ~(size_t)255; return p; };
    bf16_t* xb   = (bf16_t*)alloc((size_t)ROWS * DE * 2);   // 16 MB
    bf16_t* wqT  = (bf16_t*)alloc((size_t)DA * DE * 2);
    bf16_t* wkT  = (bf16_t*)alloc((size_t)DA * DE * 2);
    bf16_t* wvT  = (bf16_t*)alloc((size_t)DE * DE * 2);
    bf16_t* wlT  = (bf16_t*)alloc((size_t)DE * DE * 2);
    bf16_t* qb   = (bf16_t*)alloc((size_t)ROWS * DA * 2);   // 4 MB
    bf16_t* kb   = (bf16_t*)alloc((size_t)ROWS * DA * 2);   // 4 MB
    bf16_t* vT   = (bf16_t*)alloc((size_t)ROWS * DE * 2);   // 16 MB (transposed)
    bf16_t* attn = (bf16_t*)alloc((size_t)ROWS * DE * 2);   // 16 MB
    float*  h    = (float*) alloc((size_t)ROWS * DE * 4);   // 32 MB
    float*  sums = (float*) alloc((size_t)2 * DE * 4);
    float*  ss   = (float*) alloc((size_t)2 * DE * 4);

    // conversions
    k_cvt_x <<<ROWS * DE / (256 * 4), 256, 0, stream>>>(x, xb);
    k_cvt_wT<<<(DE * DA + 255) / 256, 256, 0, stream>>>(Wq, wqT, DA);
    k_cvt_wT<<<(DE * DA + 255) / 256, 256, 0, stream>>>(Wk, wkT, DA);
    k_cvt_wT<<<(DE * DE + 255) / 256, 256, 0, stream>>>(Wv, wvT, DE);
    k_cvt_wT<<<(DE * DE + 255) / 256, 256, 0, stream>>>(Wl, wlT, DE);
    (void)hipMemsetAsync(sums, 0, 2 * DE * sizeof(float), stream);

    // QKV projection: (ROWS/16) row tiles * 24 col tiles, 8 waves/block
    k_qkv<<<(ROWS / 16) * 24 / 8, 256, 0, stream>>>(xb, wqT, wkT, wvT, bq, bk, bv,
                                                    qb, kb, vT);
    // flash attention: B * (N/16) tiles, 8 waves/block
    k_attn<<<BB * (NN / 16) / 8, 256, 0, stream>>>(qb, kb, vT, attn);
    // final linear + BN stats
    k_lin<<<(ROWS / 16) * (DE / 16) / 8, 256, 0, stream>>>(attn, wlT, bl, h, sums);
    k_stats<<<1, 256, 0, stream>>>(sums, gamma, beta, ss);
    k_bnres<<<ROWS * DE / (256 * 4), 256, 0, stream>>>(h, x, ss, y);
}
